// HeteroSAGE_51711406244228
// MI455X (gfx1250) — matrix operand order repaired and checked by reading, not compiled, not run on hardware
//
#include <hip/hip_runtime.h>

typedef __attribute__((ext_vector_type(16))) _Float16 v16h;
typedef __attribute__((ext_vector_type(8)))  float    v8f;

#define NUSER 100000
#define NITEM 100000
#define NEDGE 500000
#define DIN 128
#define DH  256
#define DOUT 128

// ---------------- degree count ----------------
__global__ void deg_kernel(const int* __restrict__ dst, int E, float* __restrict__ deg) {
    int t = blockIdx.x * blockDim.x + threadIdx.x;
    if (t < E) atomicAdd(&deg[dst[t]], 1.0f);
}

// ---------------- f32 -> f16 convert ----------------
__global__ void cvt_f16(const float* __restrict__ src, _Float16* __restrict__ dst, long n) {
    long t = blockIdx.x * (long)blockDim.x + threadIdx.x;
    if (t < n) dst[t] = (_Float16)src[t];
}

// ---------------- mean + convert to f16 ----------------
__global__ void mean_cvt(const float* __restrict__ agg, const float* __restrict__ deg,
                         _Float16* __restrict__ dst, long n, int D) {
    long t = blockIdx.x * (long)blockDim.x + threadIdx.x;
    if (t >= n) return;
    float dg = deg[t / D];
    dst[t] = (_Float16)(agg[t] / fmaxf(dg, 1.0f));
}

// ---------------- edge scatter-add, f32 features, D=128 (one wave per edge) ----------------
__global__ void scatter_add_f32(const float* __restrict__ feat,
                                const int* __restrict__ src, const int* __restrict__ dst,
                                int E, float* __restrict__ agg) {
    int t = blockIdx.x * blockDim.x + threadIdx.x;
    int e = t >> 5, lane = t & 31;
    if (e >= E) return;
    int s = src[e], d = dst[e];
    const float4 v = *reinterpret_cast<const float4*>(feat + (size_t)s * DIN + lane * 4);
    float* base = agg + (size_t)d * DIN + lane * 4;
    atomicAdd(base + 0, v.x);
    atomicAdd(base + 1, v.y);
    atomicAdd(base + 2, v.z);
    atomicAdd(base + 3, v.w);
}

// ---------------- edge scatter-add, f16 features, D=256 (one wave per edge) ----------------
__global__ void scatter_add_f16(const _Float16* __restrict__ feat,
                                const int* __restrict__ src, const int* __restrict__ dst,
                                int E, float* __restrict__ agg) {
    int t = blockIdx.x * blockDim.x + threadIdx.x;
    int e = t >> 5, lane = t & 31;
    if (e >= E) return;
    int s = src[e], d = dst[e];
    union { float4 f4; _Float16 h[8]; } u;
    u.f4 = *reinterpret_cast<const float4*>(feat + (size_t)s * DH + lane * 8);
    float* base = agg + (size_t)d * DH + lane * 8;
#pragma unroll
    for (int k = 0; k < 8; ++k) atomicAdd(base + k, (float)u.h[k]);
}

// ---------------- pack W (f32, row-major [K,N]) into WMMA B-fragment order (f16) ----------------
// packed[((nt*KT+kt)*32 + lane)*16 + i] = W[kt*32 + (lane>>4)*8 + (i&7) + (i>>3)*16][nt*16 + (lane&15)]
__global__ void pack_w(const float* __restrict__ W, _Float16* __restrict__ P, int K, int N) {
    int t = blockIdx.x * blockDim.x + threadIdx.x;
    if (t >= K * N) return;
    int KT = K >> 5;
    int i    = t & 15;
    int lane = (t >> 4) & 31;
    int tile = t >> 9;
    int kt = tile % KT;
    int nt = tile / KT;
    int k = kt * 32 + (lane >> 4) * 8 + (i & 7) + (i >> 3) * 16;
    int n = nt * 16 + (lane & 15);
    P[t] = (_Float16)W[(size_t)k * N + n];
}

// ---------------- fused SAGE layer GEMM: out = act(Aself@Bself + Aneigh@Bneigh + bias) ----------------
// One wave (32 threads) computes a 32(M) x 64(N) tile with v_wmma_f32_16x16x32_f16.
__device__ __forceinline__ v16h load_a_frag(const _Float16* __restrict__ A, int K, int m, int kt, int g) {
    v16h a;
    const _Float16* p = A + (size_t)m * K + kt * 32 + g * 8;
    reinterpret_cast<float4*>(&a)[0] = *reinterpret_cast<const float4*>(p);
    reinterpret_cast<float4*>(&a)[1] = *reinterpret_cast<const float4*>(p + 16);
    return a;
}

template <bool RELU, bool OUT16>
__global__ void sage_gemm(const _Float16* __restrict__ Aself,
                          const _Float16* __restrict__ Aneigh,
                          const _Float16* __restrict__ Bself,   // packed
                          const _Float16* __restrict__ Bneigh,  // packed
                          const float* __restrict__ bias,
                          int N, int K,
                          _Float16* __restrict__ out16,
                          float* __restrict__ out32) {
    const int lane = threadIdx.x & 31;
    const int g  = lane >> 4;
    const int ml = lane & 15;
    const int row0 = blockIdx.x * 32;      // two 16-row strips per wave
    const int n0   = blockIdx.y * 64;
    const int KT = K >> 5;
    const int m0 = row0 + ml;
    const int m1 = row0 + 16 + ml;

    v8f acc0[4] = {};
    v8f acc1[4] = {};

#pragma unroll 1
    for (int pass = 0; pass < 2; ++pass) {
        const _Float16* A = pass ? Aneigh : Aself;
        const _Float16* B = pass ? Bneigh : Bself;
        for (int kt = 0; kt < KT; ++kt) {
            v16h a0 = load_a_frag(A, K, m0, kt, g);
            v16h a1 = load_a_frag(A, K, m1, kt, g);
#pragma unroll
            for (int j = 0; j < 4; ++j) {
                int ntile = (n0 >> 4) + j;
                v16h b = *reinterpret_cast<const v16h*>(
                    B + ((size_t)(ntile * KT + kt) * 32 + lane) * 16);
                acc0[j] = __builtin_amdgcn_wmma_f32_16x16x32_f16(
                    false, a0, false, b, (short)0, acc0[j], false, false);
                acc1[j] = __builtin_amdgcn_wmma_f32_16x16x32_f16(
                    false, a1, false, b, (short)0, acc1[j], false, false);
            }
        }
    }

#pragma unroll
    for (int j = 0; j < 4; ++j) {
        int n = n0 + j * 16 + ml;
        float bv = bias[n];
#pragma unroll
        for (int r = 0; r < 8; ++r) {
            int mmA = row0 + r + 8 * g;
            int mmB = row0 + 16 + r + 8 * g;
            float vA = acc0[j][r] + bv;
            float vB = acc1[j][r] + bv;
            if (RELU) {
                vA = (vA > 0.f) ? vA : 0.01f * vA;
                vB = (vB > 0.f) ? vB : 0.01f * vB;
            }
            if (OUT16) {
                out16[(size_t)mmA * N + n] = (_Float16)vA;
                out16[(size_t)mmB * N + n] = (_Float16)vB;
            } else {
                out32[(size_t)mmA * N + n] = vA;
                out32[(size_t)mmB * N + n] = vB;
            }
        }
    }
}

extern "C" void kernel_launch(void* const* d_in, const int* in_sizes, int n_in,
                              void* d_out, int out_size, void* d_ws, size_t ws_size,
                              hipStream_t stream) {
    const float* emb_user = (const float*)d_in[0];
    const float* emb_item = (const float*)d_in[1];
    const float* W1s_ui = (const float*)d_in[2];
    const float* W1n_ui = (const float*)d_in[3];
    const float* b1_ui  = (const float*)d_in[4];
    const float* W1s_iu = (const float*)d_in[5];
    const float* W1n_iu = (const float*)d_in[6];
    const float* b1_iu  = (const float*)d_in[7];
    const float* W2s_ui = (const float*)d_in[8];
    const float* W2n_ui = (const float*)d_in[9];
    const float* b2_ui  = (const float*)d_in[10];
    const float* W2s_iu = (const float*)d_in[11];
    const float* W2n_iu = (const float*)d_in[12];
    const float* b2_iu  = (const float*)d_in[13];
    const int* e1_src = (const int*)d_in[14];
    const int* e1_dst = (const int*)d_in[15];
    const int* e2_src = (const int*)d_in[16];
    const int* e2_dst = (const int*)d_in[17];

    char* ws = (char*)d_ws;
    size_t off = 0;
    auto alloc = [&](size_t bytes) -> void* {
        off = (off + 255) & ~(size_t)255;
        void* p = ws + off;
        off += bytes;
        return p;
    };

    float*    degI   = (float*)   alloc((size_t)NITEM * 4);
    float*    degU   = (float*)   alloc((size_t)NUSER * 4);
    _Float16* embI16 = (_Float16*)alloc((size_t)NITEM * DIN * 2);
    _Float16* embU16 = (_Float16*)alloc((size_t)NUSER * DIN * 2);
    _Float16* h1i    = (_Float16*)alloc((size_t)NITEM * DH * 2);
    _Float16* h1u    = (_Float16*)alloc((size_t)NUSER * DH * 2);
    _Float16* mean16 = (_Float16*)alloc((size_t)100000 * DH * 2);   // reused per stage
    float*    agg    = (float*)   alloc((size_t)100000 * DH * 4);   // reused per stage
    _Float16* pW1s_ui = (_Float16*)alloc((size_t)DIN * DH * 2);
    _Float16* pW1n_ui = (_Float16*)alloc((size_t)DIN * DH * 2);
    _Float16* pW1s_iu = (_Float16*)alloc((size_t)DIN * DH * 2);
    _Float16* pW1n_iu = (_Float16*)alloc((size_t)DIN * DH * 2);
    _Float16* pW2s_ui = (_Float16*)alloc((size_t)DH * DOUT * 2);
    _Float16* pW2n_ui = (_Float16*)alloc((size_t)DH * DOUT * 2);
    _Float16* pW2s_iu = (_Float16*)alloc((size_t)DH * DOUT * 2);
    _Float16* pW2n_iu = (_Float16*)alloc((size_t)DH * DOUT * 2);

    const int TPB = 256;
    const int edgeBlocks = (NEDGE * 32 + TPB - 1) / TPB;

    // degrees
    hipMemsetAsync(degI, 0, (size_t)NITEM * 4, stream);
    hipMemsetAsync(degU, 0, (size_t)NUSER * 4, stream);
    deg_kernel<<<(NEDGE + TPB - 1) / TPB, TPB, 0, stream>>>(e1_dst, NEDGE, degI);
    deg_kernel<<<(NEDGE + TPB - 1) / TPB, TPB, 0, stream>>>(e2_dst, NEDGE, degU);

    // pack weights into WMMA B layout (f16)
    int p1blocks = (DIN * DH + TPB - 1) / TPB;
    int p2blocks = (DH * DOUT + TPB - 1) / TPB;
    pack_w<<<p1blocks, TPB, 0, stream>>>(W1s_ui, pW1s_ui, DIN, DH);
    pack_w<<<p1blocks, TPB, 0, stream>>>(W1n_ui, pW1n_ui, DIN, DH);
    pack_w<<<p1blocks, TPB, 0, stream>>>(W1s_iu, pW1s_iu, DIN, DH);
    pack_w<<<p1blocks, TPB, 0, stream>>>(W1n_iu, pW1n_iu, DIN, DH);
    pack_w<<<p2blocks, TPB, 0, stream>>>(W2s_ui, pW2s_ui, DH, DOUT);
    pack_w<<<p2blocks, TPB, 0, stream>>>(W2n_ui, pW2n_ui, DH, DOUT);
    pack_w<<<p2blocks, TPB, 0, stream>>>(W2s_iu, pW2s_iu, DH, DOUT);
    pack_w<<<p2blocks, TPB, 0, stream>>>(W2n_iu, pW2n_iu, DH, DOUT);

    // embeddings to f16 (self inputs for layer 1)
    cvt_f16<<<((long)NITEM * DIN + TPB - 1) / TPB, TPB, 0, stream>>>(emb_item, embI16, (long)NITEM * DIN);
    cvt_f16<<<((long)NUSER * DIN + TPB - 1) / TPB, TPB, 0, stream>>>(emb_user, embU16, (long)NUSER * DIN);

    // ---- layer 1, item side (relation user->item over e1) ----
    hipMemsetAsync(agg, 0, (size_t)NITEM * DIN * 4, stream);
    scatter_add_f32<<<edgeBlocks, TPB, 0, stream>>>(emb_user, e1_src, e1_dst, NEDGE, agg);
    mean_cvt<<<((long)NITEM * DIN + TPB - 1) / TPB, TPB, 0, stream>>>(agg, degI, mean16, (long)NITEM * DIN, DIN);
    sage_gemm<true, true><<<dim3(NITEM / 32, DH / 64), 32, 0, stream>>>(
        embI16, mean16, pW1s_ui, pW1n_ui, b1_ui, DH, DIN, h1i, nullptr);

    // ---- layer 1, user side (relation item->user over e2) ----
    hipMemsetAsync(agg, 0, (size_t)NUSER * DIN * 4, stream);
    scatter_add_f32<<<edgeBlocks, TPB, 0, stream>>>(emb_item, e2_src, e2_dst, NEDGE, agg);
    mean_cvt<<<((long)NUSER * DIN + TPB - 1) / TPB, TPB, 0, stream>>>(agg, degU, mean16, (long)NUSER * DIN, DIN);
    sage_gemm<true, true><<<dim3(NUSER / 32, DH / 64), 32, 0, stream>>>(
        embU16, mean16, pW1s_iu, pW1n_iu, b1_iu, DH, DIN, h1u, nullptr);

    float* out_user = (float*)d_out;                        // h2_user [NUSER, DOUT]
    float* out_item = (float*)d_out + (size_t)NUSER * DOUT; // h2_item [NITEM, DOUT]

    // ---- layer 2, item side: gather h1_user over e1 ----
    hipMemsetAsync(agg, 0, (size_t)NITEM * DH * 4, stream);
    scatter_add_f16<<<edgeBlocks, TPB, 0, stream>>>(h1u, e1_src, e1_dst, NEDGE, agg);
    mean_cvt<<<((long)NITEM * DH + TPB - 1) / TPB, TPB, 0, stream>>>(agg, degI, mean16, (long)NITEM * DH, DH);
    sage_gemm<false, false><<<dim3(NITEM / 32, DOUT / 64), 32, 0, stream>>>(
        h1i, mean16, pW2s_ui, pW2n_ui, b2_ui, DOUT, DH, nullptr, out_item);

    // ---- layer 2, user side: gather h1_item over e2 ----
    hipMemsetAsync(agg, 0, (size_t)NUSER * DH * 4, stream);
    scatter_add_f16<<<edgeBlocks, TPB, 0, stream>>>(h1i, e2_src, e2_dst, NEDGE, agg);
    mean_cvt<<<((long)NUSER * DH + TPB - 1) / TPB, TPB, 0, stream>>>(agg, degU, mean16, (long)NUSER * DH, DH);
    sage_gemm<false, false><<<dim3(NUSER / 32, DOUT / 64), 32, 0, stream>>>(
        h1u, mean16, pW2s_iu, pW2n_iu, b2_iu, DOUT, DH, nullptr, out_user);

    (void)in_sizes; (void)n_in; (void)out_size; (void)ws_size;
}
